// RNN_20040317403424
// MI455X (gfx1250) — matrix operand order repaired and checked by reading, not compiled, not run on hardware
//
#include <hip/hip_runtime.h>

#define HSZ 64
#define TSTEPS 512
#define BT 64        // threads per block = 2 waves
#define WAVES 2

typedef __attribute__((ext_vector_type(16))) _Float16 v16h;
typedef __attribute__((ext_vector_type(8)))  _Float16 v8h;
typedef __attribute__((ext_vector_type(8)))  float    v8f;

__device__ __forceinline__ float fast_tanh(float x) {
#if __has_builtin(__builtin_amdgcn_tanhf)
  return __builtin_amdgcn_tanhf(x);
#else
  float e = __expf(2.0f * x);
  return 1.0f - 2.0f / (e + 1.0f);
#endif
}

__device__ __forceinline__ v16h cat8(v8h lo, v8h hi) {
  return __builtin_shufflevector(lo, hi, 0,1,2,3,4,5,6,7,8,9,10,11,12,13,14,15);
}

// A-fragment (16x32 f16) from row-major h[16][64] in LDS.
// lane group g=0: K=kk+0..7 then kk+16..23 ; g=1: kk+8..15 then kk+24..31
__device__ __forceinline__ v16h ldsA(const _Float16* h, int kk, int m, int g) {
  const _Float16* p = h + m * 64 + kk + g * 8;
  v8h lo = *(const v8h*)p;
  v8h hi = *(const v8h*)(p + 16);
  return cat8(lo, hi);
}

// B-fragment (32x16 f16, B[k][n] = W[n][k]) from row-major W[64][ldk] in LDS.
// lanes 0-15: K=kk+0..7 then kk+16..23, col n=lane ; lanes 16-31: K=kk+8..15 then kk+24..31, col n=lane-16
__device__ __forceinline__ v16h ldsB(const _Float16* W, int ldk, int nt, int kk, int nl, int g) {
  const _Float16* p = W + (nt * 16 + nl) * ldk + kk + g * 8;
  v8h lo = *(const v8h*)p;
  v8h hi = *(const v8h*)(p + 16);
  return cat8(lo, hi);
}

__device__ __forceinline__ v8f wmma32(v16h a, v16h b, v8f c) {
  return __builtin_amdgcn_wmma_f32_16x16x32_f16(false, a, false, b, (short)0, c, false, false);
}

__global__ __launch_bounds__(BT) void rnn_fused(
    const float* __restrict__ x,
    const float* __restrict__ w_ih0, const float* __restrict__ w_hh0,
    const float* __restrict__ b_ih0, const float* __restrict__ b_hh0,
    const float* __restrict__ w_ih1, const float* __restrict__ w_hh1,
    const float* __restrict__ b_ih1, const float* __restrict__ b_hh1,
    const float* __restrict__ w_ih2, const float* __restrict__ w_hh2,
    const float* __restrict__ b_ih2, const float* __restrict__ b_hh2,
    const float* __restrict__ w1, const float* __restrict__ bf1,
    const float* __restrict__ w2, const float* __restrict__ bf2,
    const float* __restrict__ w3, const float* __restrict__ bf3,
    float* __restrict__ out)
{
  __shared__ _Float16 sWih0[64 * 32];        // padded: cols 4..31 zero
  __shared__ _Float16 sW[5][64 * 64];        // Whh0, Wih1, Whh1, Wih2, Whh2
  __shared__ float    sBias[3][64];
  __shared__ _Float16 sH[WAVES][3 * 16 * 64];

  const int tid = threadIdx.x;

  // ---- stage weights to LDS (f32 -> f16) ----
  for (int i = tid; i < 64 * 32; i += BT) {
    int n = i >> 5, k = i & 31;
    sWih0[i] = (k < 4) ? (_Float16)w_ih0[n * 4 + k] : (_Float16)0.0f;
  }
  const float* srcs[5] = { w_hh0, w_ih1, w_hh1, w_ih2, w_hh2 };
  for (int mm = 0; mm < 5; ++mm)
    for (int i = tid; i < 64 * 64; i += BT)
      sW[mm][i] = (_Float16)srcs[mm][i];
  for (int i = tid; i < 64; i += BT) {
    sBias[0][i] = b_ih0[i] + b_hh0[i];
    sBias[1][i] = b_ih1[i] + b_hh1[i];
    sBias[2][i] = b_ih2[i] + b_hh2[i];
  }
  for (int w = 0; w < WAVES; ++w)
    for (int i = tid; i < 3 * 16 * 64; i += BT)
      sH[w][i] = (_Float16)0.0f;
  __syncthreads();

  const int wave = tid >> 5;
  const int lane = tid & 31;
  const int nl = lane & 15;       // output column within n-tile (B/C/D)
  const int g  = lane >> 4;       // lane group
  const int m  = nl;              // A-matrix row for this lane
  const int b0 = (blockIdx.x * WAVES + wave) * 16;

  _Float16* h0 = &sH[wave][0];
  _Float16* h1 = &sH[wave][16 * 64];
  _Float16* h2 = &sH[wave][2 * 16 * 64];
  const int hoff = (8 * g) * 64 + nl;   // store base: rows r+8g, col nt*16+nl via imm offsets

  // ---- hoist weight B-fragments into VGPRs (held across the whole t-loop) ----
  v16h B0[4], W0h[4][2], W1i[4][2], W1h[4][2], W2i[4][2], W2h[4][2];
#pragma unroll
  for (int nt = 0; nt < 4; ++nt) {
    B0[nt] = ldsB(sWih0, 32, nt, 0, nl, g);
#pragma unroll
    for (int kk = 0; kk < 2; ++kk) {
      W0h[nt][kk] = ldsB(sW[0], 64, nt, kk * 32, nl, g);
      W1i[nt][kk] = ldsB(sW[1], 64, nt, kk * 32, nl, g);
      W1h[nt][kk] = ldsB(sW[2], 64, nt, kk * 32, nl, g);
      W2i[nt][kk] = ldsB(sW[3], 64, nt, kk * 32, nl, g);
      W2h[nt][kk] = ldsB(sW[4], 64, nt, kk * 32, nl, g);
    }
  }
  float bias0[4], bias1[4], bias2[4];
#pragma unroll
  for (int nt = 0; nt < 4; ++nt) {
    bias0[nt] = sBias[0][nt * 16 + nl];
    bias1[nt] = sBias[1][nt * 16 + nl];
    bias2[nt] = sBias[2][nt * 16 + nl];
  }

  const v8f kZero = { 0.f, 0.f, 0.f, 0.f, 0.f, 0.f, 0.f, 0.f };
  const float4* xrow = (const float4*)x + (long)(b0 + m) * TSTEPS;
  float* hf = (float*)&sH[wave][0];   // f32 final-h2 buffer, reuses dead h0/h1 space at t==T-1

  // x A-fragment: elements 4..15 are loop-invariant zeros
  v16h ax;
#pragma unroll
  for (int i = 0; i < 16; ++i) ax[i] = (_Float16)0.0f;

  float4 xv_next = xrow[0];           // software pipeline: x load one step ahead

  for (int t = 0; t < TSTEPS; ++t) {
    const float4 xv = xv_next;
    const int tn = (t + 1 < TSTEPS) ? (t + 1) : (TSTEPS - 1);
    xv_next = xrow[tn];                               // all lanes, branchless
    __builtin_prefetch(&xrow[(t + 2 < TSTEPS) ? (t + 2) : (TSTEPS - 1)], 0, 0);
    const _Float16 hz = (_Float16)0.0f;
    ax[0] = g ? hz : (_Float16)xv.x;                  // v_cndmask, no EXEC branch
    ax[1] = g ? hz : (_Float16)xv.y;
    ax[2] = g ? hz : (_Float16)xv.z;
    ax[3] = g ? hz : (_Float16)xv.w;

    // ---------- layer 0: h0 = tanh(x W_ih0^T + h0 W_hh0^T + b) ----------
    {
      v16h a0 = ldsA(h0, 0, m, g), a1 = ldsA(h0, 32, m, g);
      v8f acc[4];
#pragma unroll
      for (int nt = 0; nt < 4; ++nt) acc[nt] = wmma32(ax, B0[nt], kZero);   // C = inline 0
#pragma unroll
      for (int nt = 0; nt < 4; ++nt) acc[nt] = wmma32(a0, W0h[nt][0], acc[nt]);
#pragma unroll
      for (int nt = 0; nt < 4; ++nt) acc[nt] = wmma32(a1, W0h[nt][1], acc[nt]);
      float tv[4][8];
#pragma unroll
      for (int nt = 0; nt < 4; ++nt)
#pragma unroll
        for (int r = 0; r < 8; ++r) tv[nt][r] = fast_tanh(acc[nt][r] + bias0[nt]);
#pragma unroll
      for (int nt = 0; nt < 4; ++nt)
#pragma unroll
        for (int r = 0; r < 8; ++r) h0[hoff + r * 64 + nt * 16] = (_Float16)tv[nt][r];
    }
    // ---------- layer 1: h1 = tanh(h0 W_ih1^T + h1 W_hh1^T + b) ----------
    {
      v16h ai0 = ldsA(h0, 0, m, g), ai1 = ldsA(h0, 32, m, g);
      v16h ah0 = ldsA(h1, 0, m, g), ah1 = ldsA(h1, 32, m, g);
      v8f acc[4];
#pragma unroll
      for (int nt = 0; nt < 4; ++nt) acc[nt] = wmma32(ai0, W1i[nt][0], kZero);
#pragma unroll
      for (int nt = 0; nt < 4; ++nt) acc[nt] = wmma32(ai1, W1i[nt][1], acc[nt]);
#pragma unroll
      for (int nt = 0; nt < 4; ++nt) acc[nt] = wmma32(ah0, W1h[nt][0], acc[nt]);
#pragma unroll
      for (int nt = 0; nt < 4; ++nt) acc[nt] = wmma32(ah1, W1h[nt][1], acc[nt]);
      float tv[4][8];
#pragma unroll
      for (int nt = 0; nt < 4; ++nt)
#pragma unroll
        for (int r = 0; r < 8; ++r) tv[nt][r] = fast_tanh(acc[nt][r] + bias1[nt]);
#pragma unroll
      for (int nt = 0; nt < 4; ++nt)
#pragma unroll
        for (int r = 0; r < 8; ++r) h1[hoff + r * 64 + nt * 16] = (_Float16)tv[nt][r];
    }
    // ---------- layer 2: h2 = tanh(h1 W_ih2^T + h2 W_hh2^T + b) ----------
    {
      v16h ai0 = ldsA(h1, 0, m, g), ai1 = ldsA(h1, 32, m, g);
      v16h ah0 = ldsA(h2, 0, m, g), ah1 = ldsA(h2, 32, m, g);
      v8f acc[4];
#pragma unroll
      for (int nt = 0; nt < 4; ++nt) acc[nt] = wmma32(ai0, W2i[nt][0], kZero);
#pragma unroll
      for (int nt = 0; nt < 4; ++nt) acc[nt] = wmma32(ai1, W2i[nt][1], acc[nt]);
#pragma unroll
      for (int nt = 0; nt < 4; ++nt) acc[nt] = wmma32(ah0, W2h[nt][0], acc[nt]);
#pragma unroll
      for (int nt = 0; nt < 4; ++nt) acc[nt] = wmma32(ah1, W2h[nt][1], acc[nt]);
      float tv[4][8];
#pragma unroll
      for (int nt = 0; nt < 4; ++nt)
#pragma unroll
        for (int r = 0; r < 8; ++r) tv[nt][r] = fast_tanh(acc[nt][r] + bias2[nt]);
#pragma unroll
      for (int nt = 0; nt < 4; ++nt)
#pragma unroll
        for (int r = 0; r < 8; ++r) h2[hoff + r * 64 + nt * 16] = (_Float16)tv[nt][r];
      if (t == TSTEPS - 1) {           // uniform branch, once
#pragma unroll
        for (int nt = 0; nt < 4; ++nt)
#pragma unroll
          for (int r = 0; r < 8; ++r) hf[hoff + r * 64 + nt * 16] = tv[nt][r];
      }
    }
  }

  // ---------- MLP head: 64 -> 32 (ReLU) -> 16 -> 1, per batch row ----------
  if (lane < 16) {
    const int row = lane;
    float a1v[32];
    for (int j = 0; j < 32; ++j) {
      float s = bf1[j];
#pragma unroll 8
      for (int k = 0; k < 64; ++k) s += hf[row * 64 + k] * w1[j * 64 + k];
      a1v[j] = fmaxf(s, 0.0f);
    }
    float a2v[16];
    for (int j = 0; j < 16; ++j) {
      float s = bf2[j];
#pragma unroll 8
      for (int k = 0; k < 32; ++k) s += a1v[k] * w2[j * 32 + k];
      a2v[j] = s;
    }
    float o = bf3[0];
#pragma unroll
    for (int k = 0; k < 16; ++k) o += a2v[k] * w3[k];
    out[b0 + row] = o;
  }
}

extern "C" void kernel_launch(void* const* d_in, const int* in_sizes, int n_in,
                              void* d_out, int out_size, void* d_ws, size_t ws_size,
                              hipStream_t stream) {
  (void)in_sizes; (void)n_in; (void)d_ws; (void)ws_size; (void)out_size;
  const float* x     = (const float*)d_in[0];
  const float* w_ih0 = (const float*)d_in[1];
  const float* w_hh0 = (const float*)d_in[2];
  const float* b_ih0 = (const float*)d_in[3];
  const float* b_hh0 = (const float*)d_in[4];
  const float* w_ih1 = (const float*)d_in[5];
  const float* w_hh1 = (const float*)d_in[6];
  const float* b_ih1 = (const float*)d_in[7];
  const float* b_hh1 = (const float*)d_in[8];
  const float* w_ih2 = (const float*)d_in[9];
  const float* w_hh2 = (const float*)d_in[10];
  const float* b_ih2 = (const float*)d_in[11];
  const float* b_hh2 = (const float*)d_in[12];
  const float* w1    = (const float*)d_in[13];
  const float* bf1   = (const float*)d_in[14];
  const float* w2    = (const float*)d_in[15];
  const float* bf2   = (const float*)d_in[16];
  const float* w3    = (const float*)d_in[17];
  const float* bf3   = (const float*)d_in[18];

  // 4096 rows / (16 rows per wave * 2 waves per block) = 128 blocks
  rnn_fused<<<dim3(128), dim3(BT), 0, stream>>>(
      x, w_ih0, w_hh0, b_ih0, b_hh0,
      w_ih1, w_hh1, b_ih1, b_hh1,
      w_ih2, w_hh2, b_ih2, b_hh2,
      w1, bf1, w2, bf2, w3, bf3,
      (float*)d_out);
}
